// ContextAwareAttention_85512798863753
// MI455X (gfx1250) — matrix operand order repaired and checked by reading, not compiled, not run on hardware
//
#include <hip/hip_runtime.h>
#include <hip/hip_bf16.h>
#include <stdint.h>
#include <cmath>

#define NN 8192
#define LL 1024
#define DD 512

typedef __attribute__((ext_vector_type(16))) __bf16 v16bf;
typedef __attribute__((ext_vector_type(8)))  float  v8f;

union Frag16 {
  v16bf v;
  uint32_t u[8];
};

__device__ __forceinline__ uint16_t f2bf(float f) {
  union { float f; uint32_t u; } c; c.f = f;
  uint32_t r = c.u + 0x7FFFu + ((c.u >> 16) & 1u);
  return (uint16_t)(r >> 16);
}

__device__ __forceinline__ v16bf frag_from(const uint16_t* p0, const uint16_t* p1) {
  Frag16 f;
  uint4 a = *(const uint4*)p0;
  uint4 b = *(const uint4*)p1;
  f.u[0] = a.x; f.u[1] = a.y; f.u[2] = a.z; f.u[3] = a.w;
  f.u[4] = b.x; f.u[5] = b.y; f.u[6] = b.z; f.u[7] = b.w;
  return f.v;
}

// A fragment: 16 rows (row0..row0+15) x K-chunk [k0, k0+32), src row-major, stride in elems.
// ISA layout: lanes 0-15 rows 0-15 K={0..7,16..23}; lanes 16-31 rows 0-15 K={8..15,24..31}.
__device__ __forceinline__ v16bf loadA(const uint16_t* p, int row0, int stride, int k0) {
  int lane = threadIdx.x & 31;
  const uint16_t* q = p + (size_t)(row0 + (lane & 15)) * stride + k0 + ((lane >> 4) << 3);
  return frag_from(q, q + 16);
}

// B fragment from B^T stored row-major [cols][K]: cols col0..col0+15, K-chunk [k0, k0+32).
// ISA layout: lane = column (mod 16); lanes 0-15 hold K={0..7,16..23}, lanes 16-31 hold K={8..15,24..31}.
__device__ __forceinline__ v16bf loadBT(const uint16_t* p, int col0, int stride, int k0) {
  int lane = threadIdx.x & 31;
  const uint16_t* q = p + (size_t)(col0 + (lane & 15)) * stride + k0 + ((lane >> 4) << 3);
  return frag_from(q, q + 16);
}

__device__ __forceinline__ v8f wmma_bf16(v16bf a, v16bf b, v8f c) {
  return __builtin_amdgcn_wmma_f32_16x16x32_bf16(false, a, false, b, (short)0, c, false, false);
}

// ---------------- conversion / transpose prep ----------------

__global__ void k_cvt_x(const float* __restrict__ x, uint16_t* __restrict__ xb) {
  int i = (blockIdx.x * 256 + threadIdx.x) * 4;
  float4 v = *(const float4*)(x + i);
  ushort4 o;
  o.x = f2bf(v.x); o.y = f2bf(v.y); o.z = f2bf(v.z); o.w = f2bf(v.w);
  *(ushort4*)(xb + i) = o;
}

// W [K][M] fp32 -> Wt [M][K] bf16
__global__ void k_transpose(const float* __restrict__ W, uint16_t* __restrict__ Wt, int K, int M) {
  int idx = blockIdx.x * 256 + threadIdx.x;
  if (idx >= K * M) return;
  int k = idx / M, m = idx - k * M;
  Wt[(size_t)m * K + k] = f2bf(W[idx]);
}

// ---------------- fused flash attention ----------------
// Block: 64 query rows, 8 waves = 4 row-groups x 2 context-halves (256 cols each).
// Stream keys in 64-blocks; online softmax; P converted C-layout->A-layout via LDS.
// B-fragment loads software-pipelined depth-2; A-fragments prefetched 1 step ahead.
__global__ void __launch_bounds__(256, 1)
k_flash(const uint16_t* __restrict__ Qb, const uint16_t* __restrict__ Kb,
        const uint16_t* __restrict__ Vt, float* __restrict__ ctx,
        uint16_t* __restrict__ ctxb) {
  __shared__ __align__(16) uint16_t Pbuf[2][4][16 * 64];

  int lane = threadIdx.x & 31, wave = threadIdx.x >> 5;
  int g = wave >> 1, h = wave & 1;
  int row0 = blockIdx.x * 64 + g * 16;
  int d0 = h * 256;
  int hi = lane >> 4, n = lane & 15;

  v8f acc[16] = {};
  float mrow[8], zrow[8];
#pragma unroll
  for (int r = 0; r < 8; ++r) { mrow[r] = -3.0e38f; zrow[r] = 0.f; }

  for (int jb = 0; jb < NN / 64; ++jb) {
    int j0 = jb * 64;
    // ---- S = Q @ K^T (pre-scaled by 1/sqrt(d) via Q), pipelined ----
    v8f S[4] = {};
    {
      v16bf a  = loadA(Qb, row0, DD, 0);
      v16bf b0 = loadBT(Kb, j0 + 0,  DD, 0);
      v16bf b1 = loadBT(Kb, j0 + 16, DD, 0);
      for (int kc = 0; kc < DD / 32; ++kc) {
        v16bf anext = a;
        if (kc + 1 < DD / 32) anext = loadA(Qb, row0, DD, (kc + 1) * 32);
#pragma unroll
        for (int ct = 0; ct < 4; ++ct) {
          v16bf bcur = (ct & 1) ? b1 : b0;
          int t = kc * 4 + ct + 2;            // flat prefetch index
          v16bf bn = bcur;
          if (t < (DD / 32) * 4) bn = loadBT(Kb, j0 + (t & 3) * 16, DD, (t >> 2) * 32);
          S[ct] = wmma_bf16(a, bcur, S[ct]);
          if (ct & 1) b1 = bn; else b0 = bn;
        }
        a = anext;
      }
    }
    // ---- online softmax stats (row = r + 8*hi, col = n within 16-lane half) ----
    float bm[8];
#pragma unroll
    for (int r = 0; r < 8; ++r)
      bm[r] = fmaxf(fmaxf(S[0][r], S[1][r]), fmaxf(S[2][r], S[3][r]));
#pragma unroll
    for (int m = 1; m < 16; m <<= 1)
#pragma unroll
      for (int r = 0; r < 8; ++r) bm[r] = fmaxf(bm[r], __shfl_xor(bm[r], m, 32));
    v8f scv;
    float ps[8];
#pragma unroll
    for (int r = 0; r < 8; ++r) {
      float mn = fmaxf(mrow[r], bm[r]);
      scv[r] = __expf(mrow[r] - mn);
      mrow[r] = mn; ps[r] = 0.f;
    }
#pragma unroll
    for (int ct = 0; ct < 4; ++ct)
#pragma unroll
      for (int r = 0; r < 8; ++r) {
        float e = __expf(S[ct][r] - mrow[r]);
        S[ct][r] = e; ps[r] += e;
      }
#pragma unroll
    for (int m = 1; m < 16; m <<= 1)
#pragma unroll
      for (int r = 0; r < 8; ++r) ps[r] += __shfl_xor(ps[r], m, 32);
#pragma unroll
    for (int r = 0; r < 8; ++r) zrow[r] = zrow[r] * scv[r] + ps[r];
#pragma unroll
    for (int dt = 0; dt < 16; ++dt) acc[dt] *= scv;

    // ---- P: C-layout -> LDS -> A-layout (double buffered, one barrier/iter) ----
    int buf = jb & 1;
    if (h == 0) {
#pragma unroll
      for (int ct = 0; ct < 4; ++ct)
#pragma unroll
        for (int r = 0; r < 8; ++r)
          Pbuf[buf][g][(r + 8 * hi) * 64 + ct * 16 + n] = f2bf(S[ct][r]);
    }
    __syncthreads();

    // ---- acc += P @ V  (B^T = Vt[d][j], contiguous), pipelined ----
    {
      v16bf a0 = loadA(&Pbuf[buf][g][0], 0, 64, 0);
      v16bf a1 = loadA(&Pbuf[buf][g][0], 0, 64, 32);
      v16bf b0 = loadBT(Vt, d0 + 0,  NN, j0);
      v16bf b1 = loadBT(Vt, d0 + 16, NN, j0);
#pragma unroll
      for (int jc = 0; jc < 2; ++jc) {
        v16bf a = jc ? a1 : a0;
#pragma unroll
        for (int dt = 0; dt < 16; ++dt) {
          v16bf bcur = (dt & 1) ? b1 : b0;
          int t = jc * 16 + dt + 2;           // flat prefetch index over both jc
          v16bf bn = bcur;
          if (t < 32) bn = loadBT(Vt, d0 + (t & 15) * 16, NN, j0 + (t >> 4) * 32);
          acc[dt] = wmma_bf16(a, bcur, acc[dt]);
          if (dt & 1) b1 = bn; else b0 = bn;
        }
      }
    }
  }

  // ---- normalize and write context (fp32 to output, bf16 copy for head) ----
  v8f rzv;
#pragma unroll
  for (int r = 0; r < 8; ++r) rzv[r] = 1.0f / zrow[r];
#pragma unroll
  for (int dt = 0; dt < 16; ++dt) {
    v8f o = acc[dt] * rzv;
    int col = d0 + dt * 16 + n;
#pragma unroll
    for (int r = 0; r < 8; ++r) {
      int row = row0 + r + 8 * hi;
      ctx[(size_t)row * DD + col] = o[r];
      ctxb[(size_t)row * DD + col] = f2bf(o[r]);
    }
  }
}

// ---------------- QKV projection: out = (xb @ W + b) * scale ----------------
// 8 waves: 4 row-groups x 2 column-halves; 64 rows x 512 cols per block. Pipelined loads.
__global__ void __launch_bounds__(256, 1)
k_proj(const uint16_t* __restrict__ xb, const uint16_t* __restrict__ Wt,
       const float* __restrict__ bias, uint16_t* __restrict__ out,
       int trans, float scale) {
  int lane = threadIdx.x & 31, wave = threadIdx.x >> 5;
  int g = wave >> 1, h = wave & 1;
  int row0 = blockIdx.x * 64 + g * 16;
  int d0 = h * 256;
  int hi = lane >> 4, n = lane & 15;

  v8f acc[16] = {};
  {
    v16bf a  = loadA(xb, row0, LL, 0);
    v16bf b0 = loadBT(Wt, d0 + 0,  LL, 0);
    v16bf b1 = loadBT(Wt, d0 + 16, LL, 0);
    for (int kc = 0; kc < LL / 32; ++kc) {
      v16bf anext = a;
      if (kc + 1 < LL / 32) anext = loadA(xb, row0, LL, (kc + 1) * 32);
#pragma unroll
      for (int dt = 0; dt < 16; ++dt) {
        v16bf bcur = (dt & 1) ? b1 : b0;
        int t = kc * 16 + dt + 2;             // flat prefetch index
        v16bf bn = bcur;
        if (t < (LL / 32) * 16) bn = loadBT(Wt, d0 + (t & 15) * 16, LL, (t >> 4) * 32);
        acc[dt] = wmma_bf16(a, bcur, acc[dt]);
        if (dt & 1) b1 = bn; else b0 = bn;
      }
      a = anext;
    }
  }
#pragma unroll
  for (int dt = 0; dt < 16; ++dt) {
    int col = d0 + dt * 16 + n;
    float bv = bias[col];
#pragma unroll
    for (int r = 0; r < 8; ++r) {
      int row = row0 + r + 8 * hi;
      float v = (acc[dt][r] + bv) * scale;
      if (trans) out[(size_t)col * NN + row] = f2bf(v);   // V^T layout [D][N]
      else       out[(size_t)row * DD + col] = f2bf(v);   // [N][D]
    }
  }
}

// ---------------- head: logits[n] = b2 + sum_d tanh(ctx@W1 + b1)[n,d] * w2[d] ----------------
__global__ void __launch_bounds__(256, 1)
k_head(const uint16_t* __restrict__ ctxb, const uint16_t* __restrict__ Wt1,
       const float* __restrict__ b1, const float* __restrict__ w2,
       const float* __restrict__ b2, float* __restrict__ logits) {
  int lane = threadIdx.x & 31, wave = threadIdx.x >> 5;
  int row0 = blockIdx.x * 128 + wave * 16;
  int hi = lane >> 4, n = lane & 15;

  float lg[8] = {};
  for (int dt = 0; dt < 32; ++dt) {
    v8f c = {};
    for (int kc = 0; kc < DD / 32; ++kc) {
      v16bf a = loadA(ctxb, row0, DD, kc * 32);
      v16bf b = loadBT(Wt1, dt * 16, DD, kc * 32);
      c = wmma_bf16(a, b, c);
    }
    int col = dt * 16 + n;
    float bb = b1[col], ww = w2[col];
#pragma unroll
    for (int r = 0; r < 8; ++r) lg[r] += tanhf(c[r] + bb) * ww;
  }
#pragma unroll
  for (int m = 1; m < 16; m <<= 1)
#pragma unroll
    for (int r = 0; r < 8; ++r) lg[r] += __shfl_xor(lg[r], m, 32);
  if (n == 0) {
    float bb2 = b2[0];
#pragma unroll
    for (int r = 0; r < 8; ++r) logits[row0 + r + 8 * hi] = lg[r] + bb2;
  }
}

// ---------------- softmax over 8192 logits -> A ----------------
__global__ void k_softmax(const float* __restrict__ logits, float* __restrict__ outA) {
  __shared__ float red[256];
  int tid = threadIdx.x;
  float m = -3.0e38f;
  for (int i = tid; i < NN; i += 256) m = fmaxf(m, logits[i]);
  red[tid] = m; __syncthreads();
  for (int s = 128; s > 0; s >>= 1) {
    if (tid < s) red[tid] = fmaxf(red[tid], red[tid + s]);
    __syncthreads();
  }
  float M = red[0]; __syncthreads();
  float s0 = 0.f;
  for (int i = tid; i < NN; i += 256) s0 += __expf(logits[i] - M);
  red[tid] = s0; __syncthreads();
  for (int s = 128; s > 0; s >>= 1) {
    if (tid < s) red[tid] += red[tid + s];
    __syncthreads();
  }
  float inv = 1.0f / red[0];
  for (int i = tid; i < NN; i += 256) outA[i] = __expf(logits[i] - M) * inv;
}

extern "C" void kernel_launch(void* const* d_in, const int* in_sizes, int n_in,
                              void* d_out, int out_size, void* d_ws, size_t ws_size,
                              hipStream_t stream) {
  const float* x  = (const float*)d_in[0];
  const float* Wq = (const float*)d_in[1];
  const float* bq = (const float*)d_in[2];
  const float* Wk = (const float*)d_in[3];
  const float* bk = (const float*)d_in[4];
  const float* Wv = (const float*)d_in[5];
  const float* bv = (const float*)d_in[6];
  const float* W1 = (const float*)d_in[7];
  const float* b1 = (const float*)d_in[8];
  const float* w2 = (const float*)d_in[9];
  const float* b2 = (const float*)d_in[10];

  char* ws = (char*)d_ws;
  size_t off = 0;
  auto alloc = [&](size_t bytes) -> char* {
    char* p = ws + off;
    off += (bytes + 255) & ~(size_t)255;
    return p;
  };
  uint16_t* xb   = (uint16_t*)alloc((size_t)NN * LL * 2);  // x in bf16
  uint16_t* WtQ  = (uint16_t*)alloc((size_t)DD * LL * 2);  // Wq^T bf16
  uint16_t* WtK  = (uint16_t*)alloc((size_t)DD * LL * 2);
  uint16_t* WtV  = (uint16_t*)alloc((size_t)DD * LL * 2);
  uint16_t* Wt1  = (uint16_t*)alloc((size_t)DD * DD * 2);  // W1^T bf16
  uint16_t* Qb   = (uint16_t*)alloc((size_t)NN * DD * 2);  // Q * 1/sqrt(d), bf16
  uint16_t* Kb   = (uint16_t*)alloc((size_t)NN * DD * 2);
  uint16_t* Vtb  = (uint16_t*)alloc((size_t)NN * DD * 2);  // V^T bf16 [D][N]
  uint16_t* ctxb = (uint16_t*)alloc((size_t)NN * DD * 2);  // context bf16
  float*    lgt  = (float*)   alloc((size_t)NN * 4);

  float* outA = (float*)d_out;      // A: [1, N]
  float* ctx  = outA + NN;          // context: [N, D] fp32

  k_cvt_x<<<(NN * LL) / 1024, 256, 0, stream>>>(x, xb);
  k_transpose<<<(LL * DD + 255) / 256, 256, 0, stream>>>(Wq, WtQ, LL, DD);
  k_transpose<<<(LL * DD + 255) / 256, 256, 0, stream>>>(Wk, WtK, LL, DD);
  k_transpose<<<(LL * DD + 255) / 256, 256, 0, stream>>>(Wv, WtV, LL, DD);
  k_transpose<<<(DD * DD + 255) / 256, 256, 0, stream>>>(W1, Wt1, DD, DD);

  const float qscale = 1.0f / sqrtf((float)DD);
  k_proj<<<NN / 64, 256, 0, stream>>>(xb, WtQ, bq, Qb, 0, qscale);
  k_proj<<<NN / 64, 256, 0, stream>>>(xb, WtK, bk, Kb, 0, 1.0f);
  k_proj<<<NN / 64, 256, 0, stream>>>(xb, WtV, bv, Vtb, 1, 1.0f);

  k_flash<<<NN / 64, 256, 0, stream>>>(Qb, Kb, Vtb, ctx, ctxb);
  k_head<<<NN / 128, 256, 0, stream>>>(ctxb, Wt1, b1, w2, b2, lgt);
  k_softmax<<<1, 256, 0, stream>>>(lgt, outA);
}
